// QuanvolutionQLSTM_65481071401055
// MI455X (gfx1250) — matrix operand order repaired and checked by reading, not compiled, not run on hardware
//
#include <hip/hip_runtime.h>

// MI455X / gfx1250, wave32. quanv collapses to a
// (B*T x 256) x (256 x 4) GEMM done with v_wmma_f32_16x16x32_f16;
// the QLSTM scan collapses to closed-form cosine products, one lane
// per batch element (rows are independent).

typedef __attribute__((ext_vector_type(16))) _Float16 v16h;
typedef __attribute__((ext_vector_type(8)))  float    v8f;

#define NBATCH 1024
#define TSTEPS 196
#define NQ 4
#define NPATCH (NBATCH * TSTEPS)   // 200704
#define NTILES (NPATCH / 16)       // 12544 (exact)
#define NCLS 10

// ---------------------------------------------------------------------------
// Kernel 1: M_w[i][j] = sum_k d_w(k) * (Ure[k,i]Ure[k,j] + Uim[k,i]Uim[k,j])
// d_w(k) = +1 if bit (3-w) of k is 0 else -1.  Output: M[w*256 + i*16 + j].
// ---------------------------------------------------------------------------
__global__ void build_m_kernel(const float* __restrict__ Ure,
                               const float* __restrict__ Uim,
                               float* __restrict__ M) {
  const int tid = threadIdx.x;           // 256 threads, tid = i*16 + j
  const int i = tid >> 4, j = tid & 15;
  float a0 = 0.f, a1 = 0.f, a2 = 0.f, a3 = 0.f;
  for (int k = 0; k < 16; ++k) {
    const float base = Ure[k * 16 + i] * Ure[k * 16 + j] +
                       Uim[k * 16 + i] * Uim[k * 16 + j];
    a0 += ((k & 8) ? -base : base);      // wire 0 <-> bit3
    a1 += ((k & 4) ? -base : base);
    a2 += ((k & 2) ? -base : base);
    a3 += ((k & 1) ? -base : base);
  }
  M[0 * 256 + tid] = a0;
  M[1 * 256 + tid] = a1;
  M[2 * 256 + tid] = a2;
  M[3 * 256 + tid] = a3;
}

// ---------------------------------------------------------------------------
// Kernel 2: quanv via WMMA.  Per 16-patch tile:
//   A (16x256 in 8 K-blocks of 32):  F_p[K] = s_i * s_j, K = i*16 + j
//   B (256x16, cols 0..3 = vecM_w):  constant, held in 8 v16h registers
//   C (16x16 f32):                   acc[rr] = Z(patch rr+8*half, wire ln)
// seq layout: seq[t*4*NBATCH + w*NBATCH + b]  (coalesced for the scan kernel)
// ---------------------------------------------------------------------------
__global__ void quanv_wmma_kernel(const float* __restrict__ x,
                                  const float* __restrict__ M,
                                  float* __restrict__ seq) {
  const int lane = threadIdx.x & 31;
  const int half = lane >> 4;            // 0: lanes 0-15, 1: lanes 16-31
  const int ln   = lane & 15;

  // B-matrix per 16-bit B layout: lane = column N, element e holds
  // K = kb*32 + 8*half + (e&7) + 16*(e>>3).  Columns >= 4 are zero pad.
  // Unconditional b128 loads from a clamped (always in-bounds) column,
  // then a lane-select to zero the pad columns -- no exec-mask churn.
  const float* mcol = M + (ln & 3) * 256 + 8 * half;
  const bool bactive = (ln < NQ);
  v16h breg[8];
#pragma unroll
  for (int kb = 0; kb < 8; ++kb) {
    const float4* src = (const float4*)(mcol + kb * 32);
    const float4 f0 = src[0], f1 = src[1], f2 = src[4], f3 = src[5];
    const float va[16] = {f0.x, f0.y, f0.z, f0.w, f1.x, f1.y, f1.z, f1.w,
                          f2.x, f2.y, f2.z, f2.w, f3.x, f3.y, f3.z, f3.w};
#pragma unroll
    for (int e = 0; e < 16; ++e)
      breg[kb][e] = (_Float16)(bactive ? va[e] : 0.0f);
  }

  const int wid = blockIdx.x * (blockDim.x >> 5) + (threadIdx.x >> 5);
  const int nw  = gridDim.x * (blockDim.x >> 5);

  for (int tile = wid; tile < NTILES; tile += nw) {
    // Lane builds the 16-amplitude RY product state for patch (tile*16 + ln).
    const int pg = tile * 16 + ln;       // pg = b*196 + t
    const int b  = pg / TSTEPS;
    const int t  = pg - b * TSTEPS;
    const int r  = t / 14;
    const int c  = t - r * 14;
    const float* img = x + b * 784 + (2 * r) * 28 + 2 * c;

    // Hide the strided 2x2 gather latency of the next tile (uniform branch).
    if (tile + nw < NTILES) {
      const int pgn = (tile + nw) * 16 + ln;
      const int bn  = pgn / TSTEPS;
      const int tn  = pgn - bn * TSTEPS;
      const int rn  = tn / 14;
      const int cn  = tn - rn * 14;
      const float* imgn = x + bn * 784 + (2 * rn) * 28 + 2 * cn;
      __builtin_prefetch(imgn, 0, 0);        // global_prefetch_b8
      __builtin_prefetch(imgn + 28, 0, 0);
    }

    const float a0 = img[0], a1 = img[1], a2 = img[28], a3 = img[29];

    float cw[4], sw[4];
    cw[0] = cosf(0.5f * a0); sw[0] = sinf(0.5f * a0);
    cw[1] = cosf(0.5f * a1); sw[1] = sinf(0.5f * a1);
    cw[2] = cosf(0.5f * a2); sw[2] = sinf(0.5f * a2);
    cw[3] = cosf(0.5f * a3); sw[3] = sinf(0.5f * a3);

    float s[16];
#pragma unroll
    for (int i = 0; i < 16; ++i)
      s[i] = ((i & 8) ? sw[0] : cw[0]) * ((i & 4) ? sw[1] : cw[1]) *
             ((i & 2) ? sw[2] : cw[2]) * ((i & 1) ? sw[3] : cw[3]);

    // sj[jj] == s[8*half + jj], built with one runtime select so every
    // register array access below is constant-indexed after unrolling.
    const float g0 = half ? sw[0] : cw[0];
    float sj[8];
#pragma unroll
    for (int jj = 0; jj < 8; ++jj)
      sj[jj] = g0 * ((jj & 4) ? sw[1] : cw[1]) * ((jj & 2) ? sw[2] : cw[2]) *
               ((jj & 1) ? sw[3] : cw[3]);

    v8f acc = {0.f, 0.f, 0.f, 0.f, 0.f, 0.f, 0.f, 0.f};
#pragma unroll
    for (int kb = 0; kb < 8; ++kb) {
      v16h a;
#pragma unroll
      for (int e = 0; e < 16; ++e) {
        // A element e (16-bit A 16x32 layout) carries F[K] with
        // row i = 2*kb + (e>>3), col j = (e&7) + 8*half.
        a[e] = (_Float16)(s[2 * kb + (e >> 3)] * sj[e & 7]);
      }
      acc = __builtin_amdgcn_wmma_f32_16x16x32_f16(
          false, a, false, breg[kb], (short)0, acc, false, false);
    }

    // C layout: VGPR rr, this lane -> (patch rr + 8*half, wire ln).
    if (ln < NQ) {
#pragma unroll
      for (int rr = 0; rr < 8; ++rr) {
        const int p2 = tile * 16 + rr + 8 * half;
        const int b2 = p2 / TSTEPS;
        const int t2 = p2 - b2 * TSTEPS;
        seq[t2 * (NQ * NBATCH) + ln * NBATCH + b2] = acc[rr];
      }
    }
  }
}

// ---------------------------------------------------------------------------
// Kernel 3: per-lane QLSTM scan + classifier head.
// qlayer(angles, rx) closed form with z_w = cos(angle_w + rx_w):
//   out = ( z1*z2*z3, z0*z1, z0*z1*z2, z0*z1*z2*z3 )
// ---------------------------------------------------------------------------
__device__ __forceinline__ float sigmoidf_(float v) {
  return 1.0f / (1.0f + expf(-v));
}

__global__ void qlstm_kernel(const float* __restrict__ seq,
                             const float* __restrict__ Wf, const float* __restrict__ bf,
                             const float* __restrict__ Wi, const float* __restrict__ bi,
                             const float* __restrict__ Wg, const float* __restrict__ bg,
                             const float* __restrict__ Wo, const float* __restrict__ bo,
                             const float* __restrict__ rxf, const float* __restrict__ rxi,
                             const float* __restrict__ rxg, const float* __restrict__ rxo,
                             const float* __restrict__ Wc, const float* __restrict__ bc,
                             float* __restrict__ out) {
  const int b = blockIdx.x * blockDim.x + threadIdx.x;
  if (b >= NBATCH) return;

  const float* Wp[4] = {Wf, Wi, Wg, Wo};
  const float* bp[4] = {bf, bi, bg, bo};
  const float* rp[4] = {rxf, rxi, rxg, rxo};

  float W[4][NQ][8], Bv[4][NQ], R[4][NQ];
#pragma unroll
  for (int g = 0; g < 4; ++g)
#pragma unroll
    for (int w = 0; w < NQ; ++w) {
      Bv[g][w] = bp[g][w];
      R[g][w]  = rp[g][w];
#pragma unroll
      for (int j = 0; j < 8; ++j) W[g][w][j] = Wp[g][w * 8 + j];
    }

  float h[NQ]   = {0.f, 0.f, 0.f, 0.f};
  float cst[NQ] = {0.f, 0.f, 0.f, 0.f};

  for (int t = 0; t < TSTEPS; ++t) {
    float comb[8];
#pragma unroll
    for (int w = 0; w < NQ; ++w) comb[w] = seq[t * (NQ * NBATCH) + w * NBATCH + b];
#pragma unroll
    for (int w = 0; w < NQ; ++w) comb[NQ + w] = h[w];

    float q[4][NQ];
#pragma unroll
    for (int g = 0; g < 4; ++g) {
      float z[NQ];
#pragma unroll
      for (int w = 0; w < NQ; ++w) {
        float ang = Bv[g][w];
#pragma unroll
        for (int j = 0; j < 8; ++j) ang += W[g][w][j] * comb[j];
        z[w] = cosf(ang + R[g][w]);
      }
      q[g][0] = z[1] * z[2] * z[3];
      q[g][1] = z[0] * z[1];
      q[g][2] = q[g][1] * z[2];
      q[g][3] = q[g][2] * z[3];
    }
#pragma unroll
    for (int w = 0; w < NQ; ++w) {
      const float fg = sigmoidf_(q[0][w]);
      const float ig = sigmoidf_(q[1][w]);
      const float gg = tanhf(q[2][w]);
      const float og = sigmoidf_(q[3][w]);
      cst[w] = fg * cst[w] + ig * gg;
      h[w]   = og * tanhf(cst[w]);
    }
  }

  // logits + log_softmax
  float l[NCLS];
  float m = -__builtin_inff();
#pragma unroll
  for (int k = 0; k < NCLS; ++k) {
    float v = bc[k];
#pragma unroll
    for (int w = 0; w < NQ; ++w) v += Wc[k * NQ + w] * h[w];
    l[k] = v;
    m = fmaxf(m, v);
  }
  float sum = 0.f;
#pragma unroll
  for (int k = 0; k < NCLS; ++k) sum += expf(l[k] - m);
  const float lse = logf(sum);
#pragma unroll
  for (int k = 0; k < NCLS; ++k) out[b * NCLS + k] = l[k] - m - lse;
}

// ---------------------------------------------------------------------------
extern "C" void kernel_launch(void* const* d_in, const int* in_sizes, int n_in,
                              void* d_out, int out_size, void* d_ws, size_t ws_size,
                              hipStream_t stream) {
  const float* x   = (const float*)d_in[0];
  const float* Ure = (const float*)d_in[1];
  const float* Uim = (const float*)d_in[2];
  const float* Wf  = (const float*)d_in[3];
  const float* bf  = (const float*)d_in[4];
  const float* Wi  = (const float*)d_in[5];
  const float* bi  = (const float*)d_in[6];
  const float* Wg  = (const float*)d_in[7];
  const float* bg  = (const float*)d_in[8];
  const float* Wo  = (const float*)d_in[9];
  const float* bo  = (const float*)d_in[10];
  const float* rxf = (const float*)d_in[11];
  const float* rxi = (const float*)d_in[12];
  const float* rxg = (const float*)d_in[13];
  const float* rxo = (const float*)d_in[14];
  const float* Wc  = (const float*)d_in[15];
  const float* bc  = (const float*)d_in[16];

  float* Mws = (float*)d_ws;             // 4*256 floats
  float* seq = Mws + 4 * 256;            // TSTEPS*4*NBATCH floats (~3.2 MB)

  build_m_kernel<<<1, 256, 0, stream>>>(Ure, Uim, Mws);
  // 196 blocks * 8 waves = 1568 waves, 8 tiles each (exact).
  quanv_wmma_kernel<<<196, 256, 0, stream>>>(x, Mws, seq);
  qlstm_kernel<<<(NBATCH + 255) / 256, 256, 0, stream>>>(
      seq, Wf, bf, Wi, bi, Wg, bg, Wo, bo, rxf, rxi, rxg, rxo, Wc, bc,
      (float*)d_out);
}
